// MotionAttn_80461917323385
// MI455X (gfx1250) — matrix operand order repaired and checked by reading, not compiled, not run on hardware
//
#include <hip/hip_runtime.h>

// ---------------------------------------------------------------------------
// MI455X (gfx1250, wave32) implementation.
// All GEMM-shaped work on v_wmma_f32_16x16x32_bf16 (bf16 in, f32 accum):
// the block is compute-bound (~0.8 TFLOP vs ~45us of HBM at 23.3 TB/s), so
// the bf16 matrix pipe is the right precision/path. LN / softmax / GELU /
// SiLU epilogues are f32 VALU fused into producer GEMMs.
// Data movement: global_load_async_to_lds_b128 (ASYNCcnt) when the toolchain
// exposes the builtin, register-staged LDS otherwise.
// ---------------------------------------------------------------------------

typedef __bf16 bf16_t;
typedef __attribute__((ext_vector_type(16))) __bf16 v16bf;
typedef __attribute__((ext_vector_type(8)))  float  v8f;

constexpr int Bsz = 32, Tsz = 512, Dsz = 1024, Hsz = 16;
constexpr int Mrows = Bsz * Tsz;      // 16384
constexpr int Dh = Dsz / Hsz;         // 64

constexpr int EPI_F32 = 0, EPI_F32_RES = 1, EPI_BF16 = 2,
              EPI_GELU_BF16 = 3, EPI_SILU_BF16 = 4;

#if defined(__AMDGCN__) && __has_builtin(__builtin_amdgcn_global_load_async_to_lds_b128) && __has_builtin(__builtin_amdgcn_s_wait_asynccnt)
#define USE_ASYNC_LDS 1
#else
#define USE_ASYNC_LDS 0
#endif

__device__ __forceinline__ bf16_t f2bf(float f) { return (bf16_t)f; }

#if USE_ASYNC_LDS
// Builtin prototype (from clang diagnostic): param0 = v4i in addrspace(1)
// ("__device__"), param1 = v4i in addrspace(3), then imm offset, imm cpol.
typedef int v4i __attribute__((vector_size(16)));
typedef __attribute__((address_space(1))) v4i* gptr_v4i;
typedef __attribute__((address_space(3))) v4i* lptr_v4i;
__device__ __forceinline__ void async_b128(const bf16_t* g, bf16_t* l) {
  __builtin_amdgcn_global_load_async_to_lds_b128((gptr_v4i)g, (lptr_v4i)l, 0, 0);
}
#endif

// A fragment (16x32, 16-bit): lane r=lane&15 holds row r; half hh=lane>>4:
// vector elems [0..7] = K 8*hh.. , elems [8..15] = K 16+8*hh..
__device__ __forceinline__ v16bf load_frag_A(const bf16_t* row, int hh) {
  union { v16bf v; uint4 q[2]; } u;
  u.q[0] = *(const uint4*)(row + 8 * hh);
  u.q[1] = *(const uint4*)(row + 16 + 8 * hh);
  return u.v;
}
// B fragment (32x16, 16-bit): lane c=lane&15 holds column c; half hh:
// vector elems [0..7] = K 8*hh.. , elems [8..15] = K 16+8*hh.. (same as A).
__device__ __forceinline__ v16bf load_frag_B(const bf16_t* row, int hh) {
  union { v16bf v; uint4 q[2]; } u;
  u.q[0] = *(const uint4*)(row + 8 * hh);
  u.q[1] = *(const uint4*)(row + 16 + 8 * hh);
  return u.v;
}
__device__ __forceinline__ v8f wmma_bf16(v16bf a, v16bf b, v8f c) {
  return __builtin_amdgcn_wmma_f32_16x16x32_bf16(false, a, false, b,
                                                 (short)0, c, false, false);
}

// ---------------------------------------------------------------------------
// Weight prep kernels
// ---------------------------------------------------------------------------

// W (K x N) f32  ->  Wt (N x K) bf16   (so GEMM B-fragments read rows of Wt)
__global__ void __launch_bounds__(256)
wt_transpose_kernel(const float* __restrict__ W, bf16_t* __restrict__ Wt,
                    int K, int N) {
  __shared__ float tile[32][33];
  const int n0 = blockIdx.x * 32, k0 = blockIdx.y * 32;
  const int tx = threadIdx.x & 31, ty = threadIdx.x >> 5;  // 32 x 8
#pragma unroll
  for (int i = 0; i < 32; i += 8)
    tile[ty + i][tx] = W[(size_t)(k0 + ty + i) * N + n0 + tx];
  __syncthreads();
#pragma unroll
  for (int i = 0; i < 32; i += 8)
    Wt[(size_t)(n0 + ty + i) * K + k0 + tx] = f2bf(tile[tx][ty + i]);
}

// conv weight (O, I, 3) f32 -> 3 stacked (O x I) bf16 mats: wt[s][o][i]
__global__ void __launch_bounds__(256)
convw_kernel(const float* __restrict__ w, bf16_t* __restrict__ wt, int D) {
  int idx = blockIdx.x * 256 + threadIdx.x;        // [0, 3*D*D)
  int s = idx / (D * D);
  int rem = idx - s * D * D;                       // o*D + i
  wt[idx] = f2bf(w[(size_t)rem * 3 + s]);
}

// DCT matrices (generated on device, exact integer angle reduction mod 4096).
// Forward Wt  = M[k][n]   = 2*cos(pi*(2n+1)k/2D)          (N rows=k, K=n)
// Inverse Wt  = Minv[n][k]= cos(ang)*scale(k)             (N rows=n, K=k)
__global__ void __launch_bounds__(256)
dct_kernel(bf16_t* __restrict__ Mt, bf16_t* __restrict__ Mit, int D) {
  int idx = blockIdx.x * 256 + threadIdx.x;        // [0, D*D)
  int k = idx / D, n = idx - k * D;
  int r = ((2 * n + 1) * k) & (4 * D - 1);         // angle/(pi/2D) mod 4D
  float ang = 3.14159265358979323846f * (float)r / (float)(2 * D);
  float c = cosf(ang);
  Mt[idx] = f2bf(2.0f * c);
  float scale = (k == 0) ? (0.5f / (float)D) : (1.0f / (float)D);
  Mit[(size_t)n * D + k] = f2bf(c * scale);
}

// ---------------------------------------------------------------------------
// LayerNorm (f32 in -> bf16 out), one row (D=1024) per 128-thread block
// ---------------------------------------------------------------------------
__global__ void __launch_bounds__(128)
ln_bf16_kernel(const float* __restrict__ x, const float* __restrict__ g,
               const float* __restrict__ b, bf16_t* __restrict__ out) {
  const size_t row = blockIdx.x;
  const float* xr = x + row * Dsz;
  bf16_t* orow = out + row * Dsz;
  float v[8], s = 0.f, s2 = 0.f;
#pragma unroll
  for (int i = 0; i < 8; i++) {
    v[i] = xr[threadIdx.x + i * 128];
    s += v[i]; s2 += v[i] * v[i];
  }
#pragma unroll
  for (int m = 16; m >= 1; m >>= 1) {
    s  += __shfl_xor(s, m, 32);
    s2 += __shfl_xor(s2, m, 32);
  }
  __shared__ float red[8];
  int wv = threadIdx.x >> 5, lane = threadIdx.x & 31;
  if (lane == 0) { red[wv] = s; red[4 + wv] = s2; }
  __syncthreads();
  s  = red[0] + red[1] + red[2] + red[3];
  s2 = red[4] + red[5] + red[6] + red[7];
  float mean = s * (1.0f / Dsz);
  float var = s2 * (1.0f / Dsz) - mean * mean;
  float inv = rsqrtf(var + 1e-5f);
#pragma unroll
  for (int i = 0; i < 8; i++) {
    int idx = threadIdx.x + i * 128;
    orow[idx] = f2bf((v[i] - mean) * inv * g[idx] + b[idx]);
  }
}

// ---------------------------------------------------------------------------
// Batched transpose: V (BH x 64 x 512) bf16 -> VT (BH x 512 x 64) bf16
// ---------------------------------------------------------------------------
__global__ void __launch_bounds__(256)
vtrans_kernel(const bf16_t* __restrict__ V, bf16_t* __restrict__ VT) {
  __shared__ bf16_t tile[32][33];
  const size_t slab = (size_t)blockIdx.x * Dh * Tsz;
  const int c0 = blockIdx.y * 32, r0 = blockIdx.z * 32;
  const int tx = threadIdx.x & 31, ty = threadIdx.x >> 5;
  const bf16_t* src = V + slab;
  bf16_t* dst = VT + slab;
#pragma unroll
  for (int i = 0; i < 32; i += 8)
    tile[ty + i][tx] = src[(r0 + ty + i) * Tsz + c0 + tx];
  __syncthreads();
#pragma unroll
  for (int i = 0; i < 32; i += 8)
    dst[(c0 + ty + i) * Dh + r0 + tx] = tile[tx][ty + i];
}

// ---------------------------------------------------------------------------
// WMMA GEMM:  out = Epi( A[M,K](bf16) @ Wt[N,K]^T (bf16) + bias ) (+ res)
// 128x256 block tile, 8 waves (2x4), each wave 64x64 -> 16 C-tiles
// (16 wmma per 16 LDS b128 fragment reads). TAPS=3 = 3-tap conv1d over M
// (per-batch zero padding), synchronous staging path.
// ---------------------------------------------------------------------------
constexpr int TM = 128, TN = 256, KC = 32, LDP = 40;  // 16B pad: conflict-free

template <int EPI, int TAPS>
__global__ void __launch_bounds__(256)
gemm_kernel(const bf16_t* __restrict__ A, const bf16_t* __restrict__ Wt,
            const float* __restrict__ bias, void* __restrict__ outp,
            const float* __restrict__ res, int M, int N, int K) {
  __shared__ __align__(16) bf16_t As[2][TM][LDP];
  __shared__ __align__(16) bf16_t Bs[2][TN][LDP];
  const int tid = threadIdx.x;
  const int lane = tid & 31, wv = tid >> 5;
  const int wm = wv & 1, wn = wv >> 1;            // 2 x 4 wave grid
  const int c = lane & 15, hh = lane >> 4;
  const int m0 = blockIdx.y * TM, n0 = blockIdx.x * TN;
  const int arow = tid >> 1, acol = (tid & 1) * 16;  // A: 128x32 over 256 thr
  const int brow = tid;                              // B: 256x32, row/thread
  const int nkc = K / KC;
  const int NC = nkc * TAPS;

  v8f zero = {};
  v8f acc[4][4];
#pragma unroll
  for (int mi = 0; mi < 4; mi++)
#pragma unroll
    for (int ni = 0; ni < 4; ni++) acc[mi][ni] = zero;

  auto compute = [&](int buf) {
    v16bf afr[4], bfr[4];
#pragma unroll
    for (int mi = 0; mi < 4; mi++)
      afr[mi] = load_frag_A(&As[buf][wm * 64 + mi * 16 + c][0], hh);
#pragma unroll
    for (int ni = 0; ni < 4; ni++)
      bfr[ni] = load_frag_B(&Bs[buf][wn * 64 + ni * 16 + c][0], hh);
#pragma unroll
    for (int mi = 0; mi < 4; mi++)
#pragma unroll
      for (int ni = 0; ni < 4; ni++)
        acc[mi][ni] = wmma_bf16(afr[mi], bfr[ni], acc[mi][ni]);
  };

#if USE_ASYNC_LDS
  if constexpr (TAPS == 1) {
    // async global->LDS pipeline (ASYNCcnt). Per iteration:
    // wait chunk cc -> barrier (also: all waves done reading other buffer)
    // -> issue chunk cc+1 into other buffer -> compute chunk cc.
    auto issue = [&](int cc, int buf) {
      const int k0 = cc * KC;
      const bf16_t* ap = A + (size_t)(m0 + arow) * K + k0 + acol;
      async_b128(ap,     &As[buf][arow][acol]);
      async_b128(ap + 8, &As[buf][arow][acol + 8]);
      const bf16_t* bp = Wt + (size_t)(n0 + brow) * K + k0;
      async_b128(bp,      &Bs[buf][brow][0]);
      async_b128(bp + 8,  &Bs[buf][brow][8]);
      async_b128(bp + 16, &Bs[buf][brow][16]);
      async_b128(bp + 24, &Bs[buf][brow][24]);
      __builtin_prefetch(ap + KC, 0, 1);   // -> global_prefetch_b8
      __builtin_prefetch(bp + KC, 0, 1);
    };
    issue(0, 0);
    for (int cc = 0; cc < NC; ++cc) {
      __builtin_amdgcn_s_wait_asynccnt(0);
      __syncthreads();
      if (cc + 1 < NC) issue(cc + 1, (cc + 1) & 1);
      compute(cc & 1);
    }
  } else
#endif
  {
    // synchronous staging (conv taps need zero-filled padding rows)
    for (int cc = 0; cc < NC; ++cc) {
      const int buf = cc & 1;
      const int s  = (TAPS == 3) ? cc / nkc : 0;
      const int k0 = (TAPS == 3) ? (cc - s * nkc) * KC : cc * KC;
      const int rowg = m0 + arow;
      const bf16_t* ap;
      bool valid = true;
      if (TAPS == 3) {
        int tp = (rowg & (Tsz - 1)) + s - 1;        // per-batch zero padding
        valid = ((unsigned)tp < (unsigned)Tsz);
        ap = A + (size_t)(rowg + s - 1) * K + k0 + acol;
      } else {
        ap = A + (size_t)rowg * K + k0 + acol;
      }
      uint4 ra0 = *(const uint4*)(ap);
      uint4 ra1 = *(const uint4*)(ap + 8);
      if (TAPS == 3 && !valid) {
        ra0 = make_uint4(0u, 0u, 0u, 0u);
        ra1 = make_uint4(0u, 0u, 0u, 0u);
      }
      const bf16_t* bp =
          Wt + (size_t)s * N * K + (size_t)(n0 + brow) * K + k0;
      uint4 rb0 = *(const uint4*)(bp);
      uint4 rb1 = *(const uint4*)(bp + 8);
      uint4 rb2 = *(const uint4*)(bp + 16);
      uint4 rb3 = *(const uint4*)(bp + 24);
      __builtin_prefetch(ap + KC, 0, 1);
      __builtin_prefetch(bp + KC, 0, 1);
      __syncthreads();                 // all waves done with this buffer
      *(uint4*)&As[buf][arow][acol]     = ra0;
      *(uint4*)&As[buf][arow][acol + 8] = ra1;
      *(uint4*)&Bs[buf][brow][0]  = rb0;
      *(uint4*)&Bs[buf][brow][8]  = rb1;
      *(uint4*)&Bs[buf][brow][16] = rb2;
      *(uint4*)&Bs[buf][brow][24] = rb3;
      __syncthreads();
      compute(buf);
    }
  }

  // epilogue (C layout: vgpr r -> row r+8*hh, lane c -> col)
#pragma unroll
  for (int mi = 0; mi < 4; mi++) {
#pragma unroll
    for (int ni = 0; ni < 4; ni++) {
      const int nn = n0 + wn * 64 + ni * 16 + c;
      const float bv = bias ? bias[nn] : 0.0f;
#pragma unroll
      for (int r = 0; r < 8; r++) {
        const int mm = m0 + wm * 64 + mi * 16 + r + 8 * hh;
        float v = acc[mi][ni][r] + bv;
        const size_t idx = (size_t)mm * N + nn;
        if constexpr (EPI == EPI_BF16) {
          ((bf16_t*)outp)[idx] = f2bf(v);
        } else if constexpr (EPI == EPI_GELU_BF16) {
          ((bf16_t*)outp)[idx] = f2bf(0.5f * v * (1.0f + erff(v * 0.70710678118f)));
        } else if constexpr (EPI == EPI_SILU_BF16) {
          ((bf16_t*)outp)[idx] = f2bf(v / (1.0f + __expf(-v)));
        } else if constexpr (EPI == EPI_F32_RES) {
          ((float*)outp)[idx] = v + res[idx];
        } else {
          ((float*)outp)[idx] = v;
        }
      }
    }
  }
}

// ---------------------------------------------------------------------------
// Channel-mixing attention. One (b,h) slab per 128-thread block (4 waves).
// Q,K,V are bf16 (B,H,64,512) (pure reinterpretation of the (B,T,D) GEMM
// output); VT is the pre-transposed V. Writes y = attn_out + x (f32).
// ---------------------------------------------------------------------------
__global__ void __launch_bounds__(128)
attn_kernel(const bf16_t* __restrict__ Q, const bf16_t* __restrict__ Km,
            const bf16_t* __restrict__ VT, const float* __restrict__ x,
            const float* __restrict__ temp, float* __restrict__ y) {
  const int bh = blockIdx.x;                    // b*H + h
  const int h = bh & (Hsz - 1);
  const size_t base = (size_t)bh * Dh * Tsz;
  const bf16_t* q  = Q  + base;
  const bf16_t* km = Km + base;
  const bf16_t* vt = VT + base;
  const int lane = threadIdx.x & 31, wv = threadIdx.x >> 5;  // 4 waves
  const int c = lane & 15, hh = lane >> 4;
  __shared__ __align__(16) bf16_t attn_s[4][16][64];

  // scores S[d,e] = sum_t Q[d,t] K[e,t]   (wave wv owns d-rows wv*16..+16)
  v8f zero = {};
  v8f acc[4] = {zero, zero, zero, zero};
  for (int tc = 0; tc < Tsz; tc += 32) {
    v16bf a = load_frag_A(q + (size_t)(wv * 16 + c) * Tsz + tc, hh);
#pragma unroll
    for (int j = 0; j < 4; j++) {
      v16bf bfr = load_frag_B(km + (size_t)(j * 16 + c) * Tsz + tc, hh);
      acc[j] = wmma_bf16(a, bfr, acc[j]);
    }
  }

  // softmax over e (64 cols) in C-fragment layout; row = r + 8*hh
  const float scale = temp[h];
#pragma unroll
  for (int r = 0; r < 8; r++) {
    float vr[4], mx = -1e30f;
#pragma unroll
    for (int j = 0; j < 4; j++) { vr[j] = acc[j][r] * scale; mx = fmaxf(mx, vr[j]); }
#pragma unroll
    for (int m = 8; m >= 1; m >>= 1) mx = fmaxf(mx, __shfl_xor(mx, m, 32));
    float sm = 0.f;
#pragma unroll
    for (int j = 0; j < 4; j++) { vr[j] = __expf(vr[j] - mx); sm += vr[j]; }
#pragma unroll
    for (int m = 8; m >= 1; m >>= 1) sm += __shfl_xor(sm, m, 32);
    const float inv = 1.0f / sm;
    const int row = r + 8 * hh;
#pragma unroll
    for (int j = 0; j < 4; j++) attn_s[wv][row][j * 16 + c] = f2bf(vr[j] * inv);
  }
  __syncthreads();

  // out[d,t] = sum_e attn[d,e] V[e,t]; B fragments from VT rows (contiguous)
  for (int tt = 0; tt < Tsz; tt += 16) {
    v8f o = zero;
#pragma unroll
    for (int e0 = 0; e0 < 64; e0 += 32) {
      v16bf a = load_frag_A(&attn_s[wv][c][e0], hh);
      v16bf bfr = load_frag_B(vt + (size_t)(tt + c) * Dh + e0, hh);
      o = wmma_bf16(a, bfr, o);
    }
#pragma unroll
    for (int r = 0; r < 8; r++) {
      const size_t idx = base + (size_t)(wv * 16 + r + 8 * hh) * Tsz + tt + c;
      y[idx] = o[r] + x[idx];                 // residual fused (flat layouts match)
    }
  }
}

// ---------------------------------------------------------------------------
// Host orchestration
// ---------------------------------------------------------------------------
extern "C" void kernel_launch(void* const* d_in, const int* in_sizes, int n_in,
                              void* d_out, int out_size, void* d_ws, size_t ws_size,
                              hipStream_t stream) {
  (void)in_sizes; (void)n_in; (void)out_size; (void)ws_size;
  const float* x      = (const float*)d_in[0];
  const float* q_w    = (const float*)d_in[1];
  const float* q_b    = (const float*)d_in[2];
  const float* k_w    = (const float*)d_in[3];
  const float* k_b    = (const float*)d_in[4];
  const float* v_w    = (const float*)d_in[5];
  const float* v_b    = (const float*)d_in[6];
  const float* temp   = (const float*)d_in[7];
  const float* ln1_w  = (const float*)d_in[8];
  const float* ln1_b  = (const float*)d_in[9];
  const float* ln2_w  = (const float*)d_in[10];
  const float* ln2_b  = (const float*)d_in[11];
  const float* ln3_w  = (const float*)d_in[12];
  const float* ln3_b  = (const float*)d_in[13];
  const float* ln4_w  = (const float*)d_in[14];
  const float* ln4_b  = (const float*)d_in[15];
  const float* lin1_w = (const float*)d_in[16];
  const float* lin1_b = (const float*)d_in[17];
  const float* lin2_w = (const float*)d_in[18];
  const float* lin2_b = (const float*)d_in[19];
  const float* m1w1   = (const float*)d_in[20];
  const float* m1b1   = (const float*)d_in[21];
  const float* m1w2   = (const float*)d_in[22];
  const float* m1b2   = (const float*)d_in[23];
  const float* m2w1   = (const float*)d_in[24];
  const float* m2b1   = (const float*)d_in[25];
  const float* m2w2   = (const float*)d_in[26];
  const float* m2b2   = (const float*)d_in[27];
  const float* c1w    = (const float*)d_in[28];
  const float* c1b    = (const float*)d_in[29];
  const float* c2w    = (const float*)d_in[30];
  const float* c2b    = (const float*)d_in[31];
  float* out = (float*)d_out;

  // workspace arena (disjoint-lifetime aliasing; peak ~334 MB)
  char* ws = (char*)d_ws;
  size_t off = 0;
  auto alloc = [&](size_t bytes) -> char* {
    char* p = ws + off;
    off += (bytes + 255) & ~(size_t)255;
    return p;
  };
  const size_t WB  = (size_t)Dsz * Dsz * sizeof(bf16_t);          // 2 MB
  const size_t ABS = (size_t)Mrows * Dsz * sizeof(bf16_t);        // 32 MB
  bf16_t* WQT  = (bf16_t*)alloc(WB);
  bf16_t* WKT  = (bf16_t*)alloc(WB);
  bf16_t* WVT  = (bf16_t*)alloc(WB);
  bf16_t* L1T  = (bf16_t*)alloc(2 * WB);
  bf16_t* L2T  = (bf16_t*)alloc(2 * WB);
  bf16_t* M1AT = (bf16_t*)alloc(2 * WB);
  bf16_t* M1BT = (bf16_t*)alloc(2 * WB);
  bf16_t* M2AT = (bf16_t*)alloc(2 * WB);
  bf16_t* M2BT = (bf16_t*)alloc(2 * WB);
  bf16_t* C1T  = (bf16_t*)alloc(3 * WB);
  bf16_t* C2T  = (bf16_t*)alloc(3 * WB);
  bf16_t* DCTM = (bf16_t*)alloc(WB);
  bf16_t* DCTI = (bf16_t*)alloc(WB);
  bf16_t* ABUF = (bf16_t*)alloc(ABS);       // ln outputs (bf16), serially reused
  bf16_t* QB   = (bf16_t*)alloc(ABS);
  bf16_t* KB   = (bf16_t*)alloc(ABS);      // KB+VB contiguous -> 64MB HBUF alias
  bf16_t* VB   = (bf16_t*)alloc(ABS);
  bf16_t* VTB  = (bf16_t*)alloc(ABS);
  float*  YBUF = (float*)alloc((size_t)Mrows * Dsz * sizeof(float));
  float*  ACC  = (float*)alloc((size_t)Mrows * Dsz * sizeof(float));
  bf16_t* DBUF  = QB;    // aliases, lifetimes disjoint (stream-ordered)
  bf16_t* HBUF  = KB;    // M x 2D bf16
  bf16_t* DBUF2 = VTB;

  const dim3 blk256(256);
  const dim3 gD(Dsz / TN, Mrows / TM);          // N=1024 GEMMs: (4,128)
  const dim3 g2D(2 * Dsz / TN, Mrows / TM);     // N=2048 GEMMs: (8,128)

  // ---- weight prep (cheap; done every launch for determinism) ----
  wt_transpose_kernel<<<dim3(Dsz / 32, Dsz / 32), blk256, 0, stream>>>(q_w, WQT, Dsz, Dsz);
  wt_transpose_kernel<<<dim3(Dsz / 32, Dsz / 32), blk256, 0, stream>>>(k_w, WKT, Dsz, Dsz);
  wt_transpose_kernel<<<dim3(Dsz / 32, Dsz / 32), blk256, 0, stream>>>(v_w, WVT, Dsz, Dsz);
  wt_transpose_kernel<<<dim3(2 * Dsz / 32, Dsz / 32), blk256, 0, stream>>>(lin1_w, L1T, Dsz, 2 * Dsz);
  wt_transpose_kernel<<<dim3(Dsz / 32, 2 * Dsz / 32), blk256, 0, stream>>>(lin2_w, L2T, 2 * Dsz, Dsz);
  wt_transpose_kernel<<<dim3(2 * Dsz / 32, Dsz / 32), blk256, 0, stream>>>(m1w1, M1AT, Dsz, 2 * Dsz);
  wt_transpose_kernel<<<dim3(Dsz / 32, 2 * Dsz / 32), blk256, 0, stream>>>(m1w2, M1BT, 2 * Dsz, Dsz);
  wt_transpose_kernel<<<dim3(2 * Dsz / 32, Dsz / 32), blk256, 0, stream>>>(m2w1, M2AT, Dsz, 2 * Dsz);
  wt_transpose_kernel<<<dim3(Dsz / 32, 2 * Dsz / 32), blk256, 0, stream>>>(m2w2, M2BT, 2 * Dsz, Dsz);
  convw_kernel<<<(3 * Dsz * Dsz) / 256, blk256, 0, stream>>>(c1w, C1T, Dsz);
  convw_kernel<<<(3 * Dsz * Dsz) / 256, blk256, 0, stream>>>(c2w, C2T, Dsz);
  dct_kernel<<<(Dsz * Dsz) / 256, blk256, 0, stream>>>(DCTM, DCTI, Dsz);

  // ---- QKV + channel attention ----
  ln_bf16_kernel<<<Mrows, 128, 0, stream>>>(x, ln1_w, ln1_b, ABUF);
  gemm_kernel<EPI_BF16, 1><<<gD, blk256, 0, stream>>>(ABUF, WQT, q_b, QB, nullptr, Mrows, Dsz, Dsz);
  gemm_kernel<EPI_BF16, 1><<<gD, blk256, 0, stream>>>(ABUF, WKT, k_b, KB, nullptr, Mrows, Dsz, Dsz);
  gemm_kernel<EPI_BF16, 1><<<gD, blk256, 0, stream>>>(ABUF, WVT, v_b, VB, nullptr, Mrows, Dsz, Dsz);
  vtrans_kernel<<<dim3(Bsz * Hsz, Tsz / 32, Dh / 32), blk256, 0, stream>>>(VB, VTB);
  attn_kernel<<<Bsz * Hsz, 128, 0, stream>>>(QB, KB, VTB, x, temp, YBUF);  // y = attn + x

  // ---- DCT -> MLP -> IDCT ----
  ln_bf16_kernel<<<Mrows, 128, 0, stream>>>(YBUF, ln2_w, ln2_b, ABUF);
  gemm_kernel<EPI_BF16, 1><<<gD, blk256, 0, stream>>>(ABUF, DCTM, nullptr, DBUF, nullptr, Mrows, Dsz, Dsz);
  gemm_kernel<EPI_GELU_BF16, 1><<<g2D, blk256, 0, stream>>>(DBUF, L1T, lin1_b, HBUF, nullptr, Mrows, 2 * Dsz, Dsz);
  gemm_kernel<EPI_BF16, 1><<<gD, blk256, 0, stream>>>(HBUF, L2T, lin2_b, DBUF2, nullptr, Mrows, Dsz, 2 * Dsz);
  gemm_kernel<EPI_BF16, 1><<<gD, blk256, 0, stream>>>(DBUF2, DCTI, nullptr, DBUF, nullptr, Mrows, Dsz, Dsz);

  // ---- dual MLP residual: y = m1 + m2 + y ----
  gemm_kernel<EPI_GELU_BF16, 1><<<g2D, blk256, 0, stream>>>(DBUF, M1AT, m1b1, HBUF, nullptr, Mrows, 2 * Dsz, Dsz);
  gemm_kernel<EPI_F32_RES, 1><<<gD, blk256, 0, stream>>>(HBUF, M1BT, m1b2, ACC, YBUF, Mrows, Dsz, 2 * Dsz);   // ACC=m1+y
  ln_bf16_kernel<<<Mrows, 128, 0, stream>>>(YBUF, ln3_w, ln3_b, ABUF);
  gemm_kernel<EPI_GELU_BF16, 1><<<g2D, blk256, 0, stream>>>(ABUF, M2AT, m2b1, HBUF, nullptr, Mrows, 2 * Dsz, Dsz);
  gemm_kernel<EPI_F32_RES, 1><<<gD, blk256, 0, stream>>>(HBUF, M2BT, m2b2, YBUF, ACC, Mrows, Dsz, 2 * Dsz);   // y=m2+ACC

  // ---- conv path (3-tap shifted GEMMs) + final residual ----
  ln_bf16_kernel<<<Mrows, 128, 0, stream>>>(YBUF, ln4_w, ln4_b, ABUF);
  gemm_kernel<EPI_SILU_BF16, 3><<<gD, blk256, 0, stream>>>(ABUF, C1T, c1b, DBUF, nullptr, Mrows, Dsz, Dsz);
  gemm_kernel<EPI_F32_RES, 3><<<gD, blk256, 0, stream>>>(DBUF, C2T, c2b, out, YBUF, Mrows, Dsz, Dsz);
}
